// TransformerEncoderBlock_13383118094782
// MI455X (gfx1250) — matrix operand order repaired and checked by reading, not compiled, not run on hardware
//
#include <hip/hip_runtime.h>
#include <hip/hip_bf16.h>

// ---------------------------------------------------------------------------
// Performer (FAVOR+) transformer encoder block for MI455X (gfx1250, wave32).
// Big GEMMs (>99% of FLOPs) run on v_wmma_f32_16x16x32_bf16 with f32 accum.
// Tiny FAVOR+ einsums (M=16, DH=64) run on VALU + LDS.
// ---------------------------------------------------------------------------

typedef __attribute__((ext_vector_type(16))) __bf16 v16bf;
typedef __attribute__((ext_vector_type(8)))  __bf16 v8bf;
typedef __attribute__((ext_vector_type(4)))  __bf16 v4bf;
typedef __attribute__((ext_vector_type(8)))  float  v8f;

#define B_   8
#define N_   4096
#define D_   768
#define H_   12
#define M_   16
#define DH_  64
#define F_   3072
#define BN_  (B_ * N_)

// ---------------------------------------------------------------------------
// WMMA fragment helpers (CDNA5 ISA 16-bit layouts, wave32)
// ---------------------------------------------------------------------------
__device__ __forceinline__ v16bf frag_cat(v8bf lo, v8bf hi) {
  return __builtin_shufflevector(lo, hi, 0, 1, 2, 3, 4, 5, 6, 7,
                                         8, 9, 10, 11, 12, 13, 14, 15);
}

// A (16x32, row-major MxK): lanes 0-15 row M=l, K: [k0..k0+7] + [k0+16..k0+23];
// lanes 16-31 row M=l, K: [k0+8..k0+15] + [k0+24..k0+31].
__device__ __forceinline__ v16bf load_frag_a(const __bf16* __restrict__ A, int lda,
                                             int row0, int k0, int lane) {
  const int hf = lane >> 4, l = lane & 15;
  const __bf16* p = A + (size_t)(row0 + l) * lda + k0 + hf * 8;
  return frag_cat(*(const v8bf*)p, *(const v8bf*)(p + 16));
}

// B (32x16, KxN) from transposed (N-major) weights: lanes 0-15 col N=l hold
// K: [k0..k0+7] + [k0+16..k0+23]; lanes 16-31 col N=l hold K: [k0+8..k0+15] + [k0+24..k0+31].
__device__ __forceinline__ v16bf load_frag_b(const __bf16* __restrict__ BT, int ldb,
                                             int col0, int k0, int lane) {
  const int hf = lane >> 4, l = lane & 15;
  const __bf16* p = BT + (size_t)(col0 + l) * ldb + k0 + hf * 8;
  return frag_cat(*(const v8bf*)p, *(const v8bf*)(p + 16));
}

__device__ __forceinline__ v8f wmma_bf16(v16bf a, v16bf b, v8f c) {
  return __builtin_amdgcn_wmma_f32_16x16x32_bf16(false, a, false, b,
                                                 (short)0, c, false, false);
}

__device__ __forceinline__ float gelu_exact(float v) {
  return 0.5f * v * (1.0f + erff(v * 0.70710678118654752f));
}

// ---------------------------------------------------------------------------
// Tiled WMMA GEMM: out(MR x Nr) = A(MR x K) @ BT(Nr x K)^T, + epilogue.
// Block = 8 waves; block tile 128x128; wave tile 32(M) x 64(N) = 8 C frags.
// All problem dims here are multiples of 128/32, so no bounds checks.
// ---------------------------------------------------------------------------
template <bool BIAS, bool RESID, bool GELU_, bool OBF16>
__global__ __launch_bounds__(256) void gemm_wmma(
    const __bf16* __restrict__ A, const __bf16* __restrict__ BT,
    const float* __restrict__ bias, const float* __restrict__ resid,
    void* __restrict__ outp, int Nr, int K) {
  const int lane = threadIdx.x & 31;
  const int wave = threadIdx.x >> 5;
  const int bm = blockIdx.y * 128 + (wave >> 1) * 32;
  const int bn = blockIdx.x * 128 + (wave & 1) * 64;

  v8f c[2][4] = {};

  for (int k0 = 0; k0 < K; k0 += 32) {
    const v16bf a0 = load_frag_a(A, K, bm,      k0, lane);
    const v16bf a1 = load_frag_a(A, K, bm + 16, k0, lane);
    const v16bf b0 = load_frag_b(BT, K, bn +  0, k0, lane);
    const v16bf b1 = load_frag_b(BT, K, bn + 16, k0, lane);
    const v16bf b2 = load_frag_b(BT, K, bn + 32, k0, lane);
    const v16bf b3 = load_frag_b(BT, K, bn + 48, k0, lane);
    c[0][0] = wmma_bf16(a0, b0, c[0][0]);
    c[0][1] = wmma_bf16(a0, b1, c[0][1]);
    c[0][2] = wmma_bf16(a0, b2, c[0][2]);
    c[0][3] = wmma_bf16(a0, b3, c[0][3]);
    c[1][0] = wmma_bf16(a1, b0, c[1][0]);
    c[1][1] = wmma_bf16(a1, b1, c[1][1]);
    c[1][2] = wmma_bf16(a1, b2, c[1][2]);
    c[1][3] = wmma_bf16(a1, b3, c[1][3]);
  }

  // C/D layout: lanes 0-15 N=l, M=g; lanes 16-31 N=l, M=8+g.
  const int l = lane & 15, hf = lane >> 4;
#pragma unroll
  for (int j = 0; j < 4; j++) {
    const int col = bn + j * 16 + l;
    float bia = 0.0f;
    if constexpr (BIAS) bia = bias[col];
#pragma unroll
    for (int i = 0; i < 2; i++) {
#pragma unroll
      for (int g = 0; g < 8; g++) {
        const int row = bm + i * 16 + hf * 8 + g;
        float v = c[i][j][g] + bia;
        if constexpr (GELU_) v = gelu_exact(v);
        if constexpr (RESID) v += resid[(size_t)row * Nr + col];
        if constexpr (OBF16)
          ((__bf16*)outp)[(size_t)row * Nr + col] = (__bf16)v;
        else
          ((float*)outp)[(size_t)row * Nr + col] = v;
      }
    }
  }
}

// ---------------------------------------------------------------------------
// f32 -> bf16 cast (vectorized x4)
// ---------------------------------------------------------------------------
__global__ __launch_bounds__(256) void cast_bf16_k(const float* __restrict__ x,
                                                   __bf16* __restrict__ y, int n4) {
  const int i = blockIdx.x * 256 + threadIdx.x;
  if (i < n4) {
    const float4 v = ((const float4*)x)[i];
    v4bf o;
    o[0] = (__bf16)v.x; o[1] = (__bf16)v.y; o[2] = (__bf16)v.z; o[3] = (__bf16)v.w;
    ((v4bf*)y)[i] = o;
  }
}

// W (K x N, f32, row-major) -> WT (N x K, bf16, row-major) via LDS tile.
__global__ __launch_bounds__(256) void transpose_cast_k(const float* __restrict__ W,
                                                        __bf16* __restrict__ WT,
                                                        int K, int N) {
  __shared__ float t[32][33];
  const int nb = blockIdx.x * 32, kb = blockIdx.y * 32;
  const int tx = threadIdx.x & 31, ty = threadIdx.x >> 5;
#pragma unroll
  for (int i = ty; i < 32; i += 8) t[i][tx] = W[(size_t)(kb + i) * N + nb + tx];
  __syncthreads();
#pragma unroll
  for (int i = ty; i < 32; i += 8)
    WT[(size_t)(nb + i) * K + kb + tx] = (__bf16)t[tx][i];
}

__global__ __launch_bounds__(256) void zero_k(float* __restrict__ p, int n) {
  const int i = blockIdx.x * 256 + threadIdx.x;
  if (i < n) p[i] = 0.0f;
}

// ---------------------------------------------------------------------------
// FAVOR+ feature map: QP[b,h,n,m] = relu(norm * q . proj[m]) + 1e-3
// For K-path also accumulates KSUM[b,h,m] = sum_n KP (LDS + global atomics).
// One block = 256 consecutive n within a single (b,h).
// ---------------------------------------------------------------------------
template <bool IS_K>
__global__ __launch_bounds__(256) void favor_features_k(
    const float* __restrict__ Q, const float* __restrict__ proj,
    float* __restrict__ QP, float* __restrict__ KSUM) {
  __shared__ float sproj[M_ * DH_];
  __shared__ float ssum[M_];
  const int tid = threadIdx.x;
  const int nchunks = N_ / 256;
  const int bh = blockIdx.x / nchunks;
  const int n = (blockIdx.x % nchunks) * 256 + tid;
  const int b = bh / H_, h = bh % H_;

  for (int i = tid; i < M_ * DH_; i += 256) sproj[i] = proj[i];
  if constexpr (IS_K) { if (tid < M_) ssum[tid] = 0.0f; }
  __syncthreads();

  const float norm = 0.35355339059327373f;  // 64^-0.25
  float q[DH_];
  const float* qs = Q + ((size_t)b * N_ + n) * D_ + h * DH_;
#pragma unroll
  for (int d = 0; d < DH_; d++) q[d] = qs[d] * norm;

  float* outp = QP + ((size_t)bh * N_ + n) * M_;
  for (int m = 0; m < M_; m++) {
    float s = 0.0f;
#pragma unroll
    for (int d = 0; d < DH_; d++) s = fmaf(q[d], sproj[m * DH_ + d], s);
    s = fmaxf(s, 0.0f) + 1e-3f;
    outp[m] = s;
    if constexpr (IS_K) atomicAdd(&ssum[m], s);
  }
  if constexpr (IS_K) {
    __syncthreads();
    if (tid < M_) atomicAdd(&KSUM[bh * M_ + tid], ssum[tid]);
  }
}

// d_inv[b,h,n] = 1 / (qp[n,:] . k_sum[b,h,:])
__global__ __launch_bounds__(256) void favor_dinv_k(const float* __restrict__ QP,
                                                    const float* __restrict__ KSUM,
                                                    float* __restrict__ DINV) {
  const int g = blockIdx.x * 256 + threadIdx.x;  // over B*H*N
  const int bh = g / N_;
  const float* qp = QP + (size_t)g * M_;
  const float* ks = KSUM + bh * M_;
  float s = 0.0f;
#pragma unroll
  for (int m = 0; m < M_; m++) s = fmaf(qp[m], ks[m], s);
  DINV[g] = 1.0f / s;
}

// context[b,h,m,d] = sum_n kp[b,h,n,m] * v[b,n,h*64+d]; one block per (b,h).
__global__ __launch_bounds__(256) void favor_context_k(const float* __restrict__ KP,
                                                       const float* __restrict__ V,
                                                       float* __restrict__ CTX) {
  const int bh = blockIdx.x, b = bh / H_, h = bh % H_;
  const int tid = threadIdx.x;
  __shared__ float skp[16 * M_];
  __shared__ float sv[16 * DH_];
  const int d = tid & 63;
  const int mb = tid >> 6;  // thread owns m = mb, mb+4, mb+8, mb+12 at column d
  float acc[4] = {0.0f, 0.0f, 0.0f, 0.0f};

  for (int n0 = 0; n0 < N_; n0 += 16) {
    for (int i = tid; i < 16 * M_; i += 256)
      skp[i] = KP[((size_t)bh * N_ + n0 + (i >> 4)) * M_ + (i & 15)];
    for (int i = tid; i < 16 * DH_; i += 256)
      sv[i] = V[((size_t)b * N_ + n0 + (i >> 6)) * D_ + h * DH_ + (i & 63)];
    __syncthreads();
#pragma unroll
    for (int nn = 0; nn < 16; nn++) {
      const float vv = sv[nn * DH_ + d];
#pragma unroll
      for (int j = 0; j < 4; j++)
        acc[j] = fmaf(skp[nn * M_ + mb + 4 * j], vv, acc[j]);
    }
    __syncthreads();
  }
#pragma unroll
  for (int j = 0; j < 4; j++)
    CTX[(size_t)bh * (M_ * DH_) + (mb + 4 * j) * DH_ + d] = acc[j];
}

// attn[b,n,h*64+d] = d_inv * (qp[n,:] . ctx[:,d]); bf16 output for Wo GEMM.
__global__ __launch_bounds__(256) void favor_attn_k(const float* __restrict__ QP,
                                                    const float* __restrict__ DINV,
                                                    const float* __restrict__ CTX,
                                                    __bf16* __restrict__ ATTNB) {
  const int bh = blockIdx.y, b = bh / H_, h = bh % H_;
  const int tid = threadIdx.x;
  __shared__ float sctx[M_ * DH_];
  for (int i = tid; i < M_ * DH_; i += 256) sctx[i] = CTX[(size_t)bh * (M_ * DH_) + i];
  __syncthreads();
  const int nl = tid >> 6, d = tid & 63;
  const int n = blockIdx.x * 4 + nl;
  const float* qp = QP + ((size_t)bh * N_ + n) * M_;
  const float di = DINV[(size_t)bh * N_ + n];
  float acc = 0.0f;
#pragma unroll
  for (int m = 0; m < M_; m++) acc = fmaf(qp[m], sctx[m * DH_ + d], acc);
  ATTNB[((size_t)b * N_ + n) * D_ + h * DH_ + d] = (__bf16)(acc * di);
}

// LayerNorm (biased variance) -> bf16 for MLP GEMM. One block per row.
__global__ __launch_bounds__(256) void layernorm_k(const float* __restrict__ X,
                                                   const float* __restrict__ gam,
                                                   const float* __restrict__ bet,
                                                   __bf16* __restrict__ Y) {
  const int row = blockIdx.x, tid = threadIdx.x;
  __shared__ float r1[256], r2[256];
  const float* x = X + (size_t)row * D_;
  float s = 0.0f, s2 = 0.0f;
  for (int i = tid; i < D_; i += 256) { const float v = x[i]; s += v; s2 = fmaf(v, v, s2); }
  r1[tid] = s; r2[tid] = s2;
  __syncthreads();
  for (int off = 128; off > 0; off >>= 1) {
    if (tid < off) { r1[tid] += r1[tid + off]; r2[tid] += r2[tid + off]; }
    __syncthreads();
  }
  const float mean = r1[0] * (1.0f / D_);
  const float var = r2[0] * (1.0f / D_) - mean * mean;
  const float rs = rsqrtf(var + 1e-5f);
  for (int i = tid; i < D_; i += 256)
    Y[(size_t)row * D_ + i] = (__bf16)((x[i] - mean) * rs * gam[i] + bet[i]);
}

// ---------------------------------------------------------------------------
// Host orchestration
// ---------------------------------------------------------------------------
extern "C" void kernel_launch(void* const* d_in, const int* in_sizes, int n_in,
                              void* d_out, int out_size, void* d_ws, size_t ws_size,
                              hipStream_t stream) {
  (void)in_sizes; (void)n_in; (void)out_size; (void)ws_size;
  const float* x    = (const float*)d_in[0];
  const float* Wq   = (const float*)d_in[1];
  const float* Wk   = (const float*)d_in[2];
  const float* Wv   = (const float*)d_in[3];
  const float* Wo   = (const float*)d_in[4];
  const float* bo   = (const float*)d_in[5];
  const float* proj = (const float*)d_in[6];
  const float* ln_g = (const float*)d_in[7];
  const float* ln_b = (const float*)d_in[8];
  const float* W1   = (const float*)d_in[9];
  const float* b1   = (const float*)d_in[10];
  const float* W2   = (const float*)d_in[11];
  const float* b2   = (const float*)d_in[12];

  char* ws = (char*)d_ws;
  size_t off = 0;
  auto alloc = [&](size_t bytes) -> void* {
    void* p = ws + off;
    off += (bytes + 255) & ~(size_t)255;
    return p;
  };

  __bf16* Xb   = (__bf16*)alloc((size_t)BN_ * D_ * 2);
  __bf16* WqT  = (__bf16*)alloc((size_t)D_ * D_ * 2);
  __bf16* WkT  = (__bf16*)alloc((size_t)D_ * D_ * 2);
  __bf16* WvT  = (__bf16*)alloc((size_t)D_ * D_ * 2);
  __bf16* WoT  = (__bf16*)alloc((size_t)D_ * D_ * 2);
  __bf16* W1T  = (__bf16*)alloc((size_t)D_ * F_ * 2);
  __bf16* W2T  = (__bf16*)alloc((size_t)F_ * D_ * 2);
  float*  Qf   = (float*)alloc((size_t)BN_ * D_ * 4);
  float*  Kf   = (float*)alloc((size_t)BN_ * D_ * 4);
  float*  Vf   = (float*)alloc((size_t)BN_ * D_ * 4);
  float*  QP   = (float*)alloc((size_t)B_ * H_ * N_ * M_ * 4);
  float*  KP   = (float*)alloc((size_t)B_ * H_ * N_ * M_ * 4);
  float*  KSUM = (float*)alloc((size_t)B_ * H_ * M_ * 4);
  float*  DINV = (float*)alloc((size_t)B_ * H_ * N_ * 4);
  float*  CTX  = (float*)alloc((size_t)B_ * H_ * M_ * DH_ * 4);
  __bf16* ATTNB = (__bf16*)alloc((size_t)BN_ * D_ * 2);
  float*  Y1   = (float*)alloc((size_t)BN_ * D_ * 4);
  __bf16* HB   = (__bf16*)alloc((size_t)BN_ * D_ * 2);
  __bf16* H1B  = (__bf16*)alloc((size_t)BN_ * F_ * 2);

  // 1) cast x -> bf16, transpose+cast all weights (N-major for B fragments)
  { const int n4 = BN_ * D_ / 4;
    cast_bf16_k<<<(n4 + 255) / 256, 256, 0, stream>>>(x, Xb, n4); }
  transpose_cast_k<<<dim3(D_ / 32, D_ / 32), 256, 0, stream>>>(Wq, WqT, D_, D_);
  transpose_cast_k<<<dim3(D_ / 32, D_ / 32), 256, 0, stream>>>(Wk, WkT, D_, D_);
  transpose_cast_k<<<dim3(D_ / 32, D_ / 32), 256, 0, stream>>>(Wv, WvT, D_, D_);
  transpose_cast_k<<<dim3(D_ / 32, D_ / 32), 256, 0, stream>>>(Wo, WoT, D_, D_);
  transpose_cast_k<<<dim3(F_ / 32, D_ / 32), 256, 0, stream>>>(W1, W1T, D_, F_);
  transpose_cast_k<<<dim3(D_ / 32, F_ / 32), 256, 0, stream>>>(W2, W2T, F_, D_);

  // 2) Q/K/V projections (WMMA bf16, f32 out)
  const dim3 gQKV(D_ / 128, BN_ / 128);
  gemm_wmma<false, false, false, false><<<gQKV, 256, 0, stream>>>(Xb, WqT, nullptr, nullptr, Qf, D_, D_);
  gemm_wmma<false, false, false, false><<<gQKV, 256, 0, stream>>>(Xb, WkT, nullptr, nullptr, Kf, D_, D_);
  gemm_wmma<false, false, false, false><<<gQKV, 256, 0, stream>>>(Xb, WvT, nullptr, nullptr, Vf, D_, D_);

  // 3) FAVOR+ (tiny einsums, VALU)
  zero_k<<<(B_ * H_ * M_ + 255) / 256, 256, 0, stream>>>(KSUM, B_ * H_ * M_);
  favor_features_k<false><<<B_ * H_ * (N_ / 256), 256, 0, stream>>>(Qf, proj, QP, nullptr);
  favor_features_k<true><<<B_ * H_ * (N_ / 256), 256, 0, stream>>>(Kf, proj, KP, KSUM);
  favor_dinv_k<<<B_ * H_ * N_ / 256, 256, 0, stream>>>(QP, KSUM, DINV);
  favor_context_k<<<B_ * H_, 256, 0, stream>>>(KP, Vf, CTX);
  favor_attn_k<<<dim3(N_ / 4, B_ * H_), 256, 0, stream>>>(QP, DINV, CTX, ATTNB);

  // 4) attn @ Wo + bo + x  (residual 1, f32)
  gemm_wmma<true, true, false, false><<<dim3(D_ / 128, BN_ / 128), 256, 0, stream>>>(
      ATTNB, WoT, bo, x, Y1, D_, D_);

  // 5) LayerNorm -> bf16
  layernorm_k<<<BN_, 256, 0, stream>>>(Y1, ln_g, ln_b, HB);

  // 6) MLP: gelu(h @ W1 + b1) (bf16 out), then @ W2 + b2 + Y1 -> d_out (f32)
  gemm_wmma<true, false, true, true><<<dim3(F_ / 128, BN_ / 128), 256, 0, stream>>>(
      HB, W1T, b1, nullptr, H1B, F_, D_);
  gemm_wmma<true, true, false, false><<<dim3(D_ / 128, BN_ / 128), 256, 0, stream>>>(
      H1B, W2T, b2, Y1, (float*)d_out, D_, F_);
}
